// AttentionHead_37357625541438
// MI455X (gfx1250) — matrix operand order repaired and checked by reading, not compiled, not run on hardware
//
#include <hip/hip_runtime.h>

// ---------------------------------------------------------------------------
// Causal "transposed" attention head for MI455X (gfx1250, wave32, WMMA).
//   out[b,t,:] = sum_{s<=t} softmax_s( (k_t . q_s) * C^-0.5 ) * v[s,:]
// with k,q,v = x@W + b. Implemented as flash-attention with Q<-k, K<-q, V<-v.
// All matmuls use v_wmma_f32_16x16x32_bf16 (fp32 accumulate).
// Softmax runs in log2 domain: scale 1/32 * log2(e) is folded into stored k,
// so exponentials are raw v_exp_f32 (= 2^x on AMD).
// ---------------------------------------------------------------------------

typedef __bf16 bf16_t;
typedef __attribute__((ext_vector_type(16))) __bf16 v16bf;
typedef __attribute__((ext_vector_type(8)))  __bf16 v8bf;
typedef __attribute__((ext_vector_type(8)))  float  v8f;

#define B_DIM 4
#define T_DIM 4096
#define C_DIM 1024
#define H_DIM 64
// C^-0.5 * log2(e) = (1/32) * 1.4426950408889634 (applied in fp32, then one
// rounding to bf16 -- same error budget as before).
#define KSCALE_L2E 0.045084220027780106f

// Workspace layout (bytes):
#define WS_WT   0                                    // 3 * 64*1024 bf16 = 393216
#define WS_KBF  (WS_WT + 3 * 64 * 1024 * 2)          // [B*T,64] bf16  = 2 MiB
#define WS_QBF  (WS_KBF + B_DIM * T_DIM * H_DIM * 2) // [B*T,64] bf16  = 2 MiB
#define WS_VT   (WS_QBF + B_DIM * T_DIM * H_DIM * 2) // [B,64,T] bf16  = 2 MiB

// Native fptrunc -> hardware bf16 convert (adjacent pairs fuse to cvt_pk).
__device__ __forceinline__ bf16_t cvt_bf16(float f) { return (bf16_t)f; }

// XOR-16 lane swap within a wave32 (ds_swizzle group-of-32: and=0x1F, xor=0x10)
__device__ __forceinline__ float xor16f(float v) {
  return __int_as_float(__builtin_amdgcn_ds_swizzle(__float_as_int(v), 0x401F));
}
__device__ __forceinline__ unsigned xor16u(unsigned v) {
  return (unsigned)__builtin_amdgcn_ds_swizzle((int)v, 0x401F);
}

// Pack two floats as bf16 pair in one u32 (lo | hi<<16).
__device__ __forceinline__ unsigned pack_bf16x2(float lo, float hi) {
  bf16_t a = cvt_bf16(lo), b = cvt_bf16(hi);
  unsigned short ua, ub;
  __builtin_memcpy(&ua, &a, 2);
  __builtin_memcpy(&ub, &b, 2);
  return (unsigned)ua | ((unsigned)ub << 16);
}

__device__ __forceinline__ bf16_t bf16_from_bits(unsigned short s) {
  bf16_t h;
  __builtin_memcpy(&h, &s, 2);
  return h;
}

__device__ __forceinline__ v8f wmma_bf16(v16bf a, v16bf b, v8f c) {
  // 8 args: (neg_a, A, neg_b, B, c_mod, C, reuse_a, reuse_b)
  return __builtin_amdgcn_wmma_f32_16x16x32_bf16(false, a, false, b, (short)0, c,
                                                 false, false);
}

__device__ __forceinline__ v16bf cat8(v8bf lo, v8bf hi) {
  return __builtin_shufflevector(lo, hi, 0, 1, 2, 3, 4, 5, 6, 7,
                                          8, 9, 10, 11, 12, 13, 14, 15);
}

// ---------------------------------------------------------------------------
// Kernel 1: transpose weights [C,H] fp32 -> [H,C] bf16 (3 matrices).
// ---------------------------------------------------------------------------
__global__ __launch_bounds__(256) void prep_wt(const float* __restrict__ Wk,
                                               const float* __restrict__ Wq,
                                               const float* __restrict__ Wv,
                                               bf16_t* __restrict__ wt) {
  int idx = blockIdx.x * 256 + threadIdx.x;  // 0 .. 3*64*1024-1
  int mat = idx >> 16;
  int rem = idx & 65535;
  int n = rem >> 10;
  int k = rem & 1023;
  const float* W = (mat == 0) ? Wk : (mat == 1) ? Wq : Wv;
  wt[idx] = cvt_bf16(W[k * H_DIM + n]);  // wt[mat][n][k]
}

// ---------------------------------------------------------------------------
// Kernel 2: projections. One wave = one 16-row tile of one matrix.
// k -> kbf (scaled 1/32*log2e), q -> qbf (both [B*T,64] row-major bf16),
// v -> vt transposed [B,64,T] bf16.
// ---------------------------------------------------------------------------
__global__ __launch_bounds__(256) void proj_kernel(
    const float* __restrict__ x, const bf16_t* __restrict__ wt,
    const float* __restrict__ bk, const float* __restrict__ bq,
    const float* __restrict__ bv, bf16_t* __restrict__ kbf,
    bf16_t* __restrict__ qbf, bf16_t* __restrict__ vt) {
  int lane = threadIdx.x & 31;
  int l15 = lane & 15;
  int g = lane >> 4;
  int wg = blockIdx.x * 8 + (threadIdx.x >> 5);  // 0..3071
  int mat = wg >> 10;                            // 0=k, 1=q, 2=v
  int row0 = (wg & 1023) << 4;                   // global row tile over B*T

  const bf16_t* wtm = wt + mat * (H_DIM * C_DIM);
  const float* xr = x + (size_t)(row0 + l15) * C_DIM;

  v8f zero = {};
  v8f acc[4] = {zero, zero, zero, zero};

  for (int k0 = 0; k0 < C_DIM; k0 += 32) {
    // A fragment: x rows (16x32).  lane l: M=l&15; K = g*16 + j
    v8f xlo = *(const v8f*)(xr + k0 + g * 16);
    v8f xhi = *(const v8f*)(xr + k0 + g * 16 + 8);
    v16bf a;
#pragma unroll
    for (int j = 0; j < 8; ++j) {
      a[j] = cvt_bf16(xlo[j]);
      a[j + 8] = cvt_bf16(xhi[j]);
    }
    // B fragments: W^T (32x16).  lane l: N=l&15; K = g*16 + j
    const bf16_t* wrow = wtm + (size_t)l15 * C_DIM + k0 + g * 16;
#pragma unroll
    for (int nt = 0; nt < 4; ++nt) {
      v16bf bfr = *(const v16bf*)(wrow + nt * 16 * C_DIM);
      acc[nt] = wmma_bf16(a, bfr, acc[nt]);
    }
  }

  const float* bias = (mat == 0) ? bk : (mat == 1) ? bq : bv;
#pragma unroll
  for (int nt = 0; nt < 4; ++nt) {
    int n = nt * 16 + l15;
    float bb = bias[n];
#pragma unroll
    for (int r = 0; r < 8; ++r) {
      int mg = row0 + 8 * g + r;  // C/D layout: M = 8g + r
      float val = acc[nt][r] + bb;
      if (mat == 0) {
        kbf[(size_t)mg * H_DIM + n] = cvt_bf16(val * KSCALE_L2E);
      } else if (mat == 1) {
        qbf[(size_t)mg * H_DIM + n] = cvt_bf16(val);
      } else {
        int b = mg >> 12, t = mg & (T_DIM - 1);
        vt[((size_t)b * H_DIM + n) * T_DIM + t] = cvt_bf16(val);
      }
    }
  }
}

// ---------------------------------------------------------------------------
// Kernel 3: flash attention. One wave = one (b, 16-wide t-tile).
// Scores transposed: S'[s,t] = q_s . (k_t * scale * log2e); softmax over s
// (= rows M) -> column reduce = in-lane 8 + one XOR-16 swizzle.
// Exponentials are exp2 (v_exp_f32). Output accumulated transposed:
// O'[h,t] = V^T x P'.
// ---------------------------------------------------------------------------
__global__ __launch_bounds__(256) void attn_kernel(
    const bf16_t* __restrict__ kbf, const bf16_t* __restrict__ qbf,
    const bf16_t* __restrict__ vt, float* __restrict__ out) {
  int lane = threadIdx.x & 31;
  int l15 = lane & 15;
  int g = lane >> 4;
  int tileId = blockIdx.x * 8 + (threadIdx.x >> 5);  // 0..1023
  int b = tileId >> 8;
  int t0 = (tileId & 255) << 4;

  const bf16_t* kb = kbf + (size_t)b * T_DIM * H_DIM;
  const bf16_t* qb = qbf + (size_t)b * T_DIM * H_DIM;
  const bf16_t* vb = vt + (size_t)b * H_DIM * T_DIM;

  // Fixed B-operands for scores: k^T tile (h x t). lane: N=t0+l15, K=h=g*8+j / 16+g*8+(j-8)
  const bf16_t* krow = kb + (size_t)(t0 + l15) * H_DIM + g * 8;
  v16bf bk0 = cat8(*(const v8bf*)(krow), *(const v8bf*)(krow + 16));       // h in [0,32)
  v16bf bk1 = cat8(*(const v8bf*)(krow + 32), *(const v8bf*)(krow + 48));  // h in [32,64)

  v8f zero = {};
  v8f acc[4] = {zero, zero, zero, zero};  // O'[h,t], 4 h-tiles of 16
  float m_run = -__builtin_inff();
  float l_run = 0.f;
  int tcol = t0 + l15;
  int s_end = t0 + 15;

  for (int sb = 0; sb <= s_end; sb += 32) {
    // A fragments: q rows (s x h). lane: M = sb(+16)+l15
    const bf16_t* q0 = qb + (size_t)(sb + l15) * H_DIM;
    const bf16_t* q1 = q0 + 16 * H_DIM;
    v16bf a00 = cat8(*(const v8bf*)(q0 + g * 8), *(const v8bf*)(q0 + 16 + g * 8));
    v16bf a01 = cat8(*(const v8bf*)(q0 + 32 + g * 8), *(const v8bf*)(q0 + 48 + g * 8));
    v16bf a10 = cat8(*(const v8bf*)(q1 + g * 8), *(const v8bf*)(q1 + 16 + g * 8));
    v16bf a11 = cat8(*(const v8bf*)(q1 + 32 + g * 8), *(const v8bf*)(q1 + 48 + g * 8));

    v8f st0 = wmma_bf16(a01, bk1, wmma_bf16(a00, bk0, zero));  // s in [sb,sb+16)
    v8f st1 = wmma_bf16(a11, bk1, wmma_bf16(a10, bk0, zero));  // s in [sb+16,sb+32)

    // causal mask (keep s <= t) + per-column tile max (log2-domain scores)
    float cm = -__builtin_inff();
#pragma unroll
    for (int r = 0; r < 8; ++r) {
      int srow = sb + 8 * g + r;  // C/D layout: M = 8g + r
      float v0 = (srow <= tcol) ? st0[r] : -__builtin_inff();
      float v1 = (srow + 16 <= tcol) ? st1[r] : -__builtin_inff();
      st0[r] = v0;
      st1[r] = v1;
      cm = fmaxf(cm, fmaxf(v0, v1));
    }
    cm = fmaxf(cm, xor16f(cm));  // column spans lane pair (l, l^16)
    float m_new = fmaxf(m_run, cm);
    float corr = __builtin_amdgcn_exp2f(m_run - m_new);

    float p0[8], p1[8];
    float ts = 0.f;
#pragma unroll
    for (int r = 0; r < 8; ++r) {
      p0[r] = __builtin_amdgcn_exp2f(st0[r] - m_new);  // masked (-inf) -> 0
      p1[r] = __builtin_amdgcn_exp2f(st1[r] - m_new);
      ts += p0[r] + p1[r];
    }
    ts += xor16f(ts);
    l_run = l_run * corr + ts;
    m_run = m_new;
#pragma unroll
    for (int ht = 0; ht < 4; ++ht) acc[ht] *= corr;

    // Reshape P' (two C/D tiles) -> B-operand (K=s over 32, N=t).
    // B elem j: K = g*16+j.  g=0 needs tile0 rows 0..15, g=1 needs tile1 rows.
    // Pack (p0,p1) as bf16x2 before the swizzle: 8 ds_swizzles instead of 16.
    v16bf pb;
#pragma unroll
    for (int j = 0; j < 8; ++j) {
      unsigned own = pack_bf16x2(p0[j], p1[j]);  // lo=tile0 row, hi=tile1 row
      unsigned oth = xor16u(own);
      // pb[j]   (K rows 0..7 of my tile):  g=0 -> my p0[j];     g=1 -> partner p1[j]
      // pb[j+8] (K rows 8..15 of my tile): g=0 -> partner p0[j]; g=1 -> my p1[j]
      unsigned short e0 = g ? (unsigned short)(oth >> 16) : (unsigned short)own;
      unsigned short e1 = g ? (unsigned short)(own >> 16) : (unsigned short)oth;
      pb[j] = bf16_from_bits(e0);
      pb[j + 8] = bf16_from_bits(e1);
    }

    // O'[h,t] += V^T x P'.  A elem j: M=h=ht*16+l15, K = g*16+j
#pragma unroll
    for (int ht = 0; ht < 4; ++ht) {
      const bf16_t* vrow = vb + (size_t)(ht * 16 + l15) * T_DIM + sb;
      v16bf av = cat8(*(const v8bf*)(vrow + g * 16), *(const v8bf*)(vrow + g * 16 + 8));
      acc[ht] = wmma_bf16(av, pb, acc[ht]);
    }
  }

  float inv = 1.f / l_run;
#pragma unroll
  for (int ht = 0; ht < 4; ++ht) {
#pragma unroll
    for (int r = 0; r < 8; ++r) {
      int h = ht * 16 + 8 * g + r;  // O' layout: M=h, N=t
      out[((size_t)b * T_DIM + t0 + l15) * H_DIM + h] = acc[ht][r] * inv;
    }
  }
}

// ---------------------------------------------------------------------------
extern "C" void kernel_launch(void* const* d_in, const int* in_sizes, int n_in,
                              void* d_out, int out_size, void* d_ws,
                              size_t ws_size, hipStream_t stream) {
  (void)in_sizes; (void)n_in; (void)out_size; (void)ws_size;
  const float* x  = (const float*)d_in[0];
  const float* Wk = (const float*)d_in[1];
  const float* bk = (const float*)d_in[2];
  const float* Wq = (const float*)d_in[3];
  const float* bq = (const float*)d_in[4];
  const float* Wv = (const float*)d_in[5];
  const float* bv = (const float*)d_in[6];
  float* out = (float*)d_out;

  char* ws = (char*)d_ws;
  bf16_t* wt  = (bf16_t*)(ws + WS_WT);
  bf16_t* kbf = (bf16_t*)(ws + WS_KBF);
  bf16_t* qbf = (bf16_t*)(ws + WS_QBF);
  bf16_t* vtb = (bf16_t*)(ws + WS_VT);

  prep_wt<<<768, 256, 0, stream>>>(Wk, Wq, Wv, wt);                 // 3*64*1024 elems
  proj_kernel<<<384, 256, 0, stream>>>(x, wt, bk, bq, bv, kbf, qbf, vtb);  // 3072 waves
  attn_kernel<<<128, 256, 0, stream>>>(kbf, qbf, vtb, out);         // 1024 waves
}
